// LinearMultiheadAttention_28149215658103
// MI455X (gfx1250) — matrix operand order repaired and checked by reading, not compiled, not run on hardware
//
#include <hip/hip_runtime.h>
#include <hip/hip_bf16.h>

typedef __attribute__((ext_vector_type(16))) _Float16 v16h;
typedef __attribute__((ext_vector_type(8)))  _Float16 v8h;
typedef __attribute__((ext_vector_type(8)))  float    v8f;
typedef __attribute__((ext_vector_type(4)))  float    f4v;
typedef __attribute__((ext_vector_type(4)))  unsigned int v4u;
typedef __attribute__((ext_vector_type(8)))  int      v8i;
typedef __attribute__((ext_vector_type(4)))  int      v4i;

union F16x16 { v16h v; v8h h[2]; };

// ---- WMMA fragment helpers (CDNA5 16x16x32 f16 -> f32) ----------------------
// A (16x32): lane l holds row (l&15); halves 0..7 = K = (l>>4)*8 .. +7,
//            halves 8..15 = K = 16 + (l>>4)*8 .. +7.  Tile stored [M][K].
__device__ __forceinline__ v16h frag_a(const _Float16* base, int lda, int lane) {
  const _Float16* p = base + (lane & 15) * lda + ((lane >> 4) << 3);
  F16x16 f;
  f.h[0] = *(const v8h*)(p);
  f.h[1] = *(const v8h*)(p + 16);
  return f.v;
}
// B (32x16): lane l holds col (l&15); halves 0..7 = K = (l>>4)*8 .. +7,
// halves 8..15 = K = 16 + (l>>4)*8 .. +7. Tile stored [N][K] (K contiguous).
__device__ __forceinline__ v16h frag_b(const _Float16* base, int ldb, int lane) {
  const _Float16* p = base + (lane & 15) * ldb + ((lane >> 4) << 3);
  F16x16 f;
  f.h[0] = *(const v8h*)(p);
  f.h[1] = *(const v8h*)(p + 16);
  return f.v;
}
__device__ __forceinline__ v8f wmma_f16(v16h a, v16h b, v8f c) {
  return __builtin_amdgcn_wmma_f32_16x16x32_f16(false, a, false, b, (short)0, c,
                                                false, false);
}
__device__ __forceinline__ v8f vzero8() {
  v8f z;
#pragma unroll
  for (int i = 0; i < 8; ++i) z[i] = 0.0f;
  return z;
}

// ---- TDM: async tensor load of a 128x64 f16 tile into LDS with +8-half row
// pad (LDS row stride 72 halves). D# per CDNA5 ISA §8 (2D, groups 2/3 zero).
__device__ __forceinline__ void tdm_load_tile_128x64(const void* gptr,
                                                     unsigned lds_off) {
  const unsigned long long ga = (unsigned long long)(size_t)gptr;
  v4u g0;
  g0[0] = 1u;                                   // count=1, user mode
  g0[1] = lds_off;                              // lds_addr (bytes)
  g0[2] = (unsigned)(ga & 0xffffffffu);         // global_addr[31:0]
  g0[3] = (unsigned)((ga >> 32) & 0x01ffffffu)  // global_addr[56:32]
          | (2u << 30);                         // type=2 ("image")
  v8i g1;
  g1[0] = (1 << 16)      // data_size = 1 -> 2 bytes
        | (1 << 20)      // pad_enable
        | (4 << 22)      // pad_interval code 4 -> every 32 DWORDs (128B row)
        | (3 << 25);     // pad_amount  code 3 -> 4 DWORDs (8 halves)
  g1[1] = (64  << 16);   // tensor_dim0[15:0]=64 (atomic_barrier_addr=0)
  g1[2] = (128 << 16);   // tensor_dim0[31:16]=0 | tensor_dim1[15:0]=128
  g1[3] = (64  << 16);   // tensor_dim1[31:16]=0 | tile_dim0=64
  g1[4] = 128;           // tile_dim1=128, tile_dim2=0
  g1[5] = 64;            // tensor_dim0_stride = 64 elements
  g1[6] = 0;
  g1[7] = 0;
  v4i z4;
  z4[0] = z4[1] = z4[2] = z4[3] = 0;
  v8i z8;
#pragma unroll
  for (int i = 0; i < 8; ++i) z8[i] = 0;
  // 6-arg form on this toolchain: (g0, g1, g2, g3, extra, cpol)
  __builtin_amdgcn_tensor_load_to_lds(g0, g1, z4, z4, z8, 0);
}

// ---- Kernel 1: fused QKV projection, f32 x * f32 W^T -> phi -> f16 head-major
// grid = (N/128, M/128, 3)   z: 0=q(phi) 1=k(phi) 2=v
__global__ __launch_bounds__(256) void qkv_gemm(
    const float* __restrict__ x,
    const float* __restrict__ Wq, const float* __restrict__ Wk,
    const float* __restrict__ Wv,
    const float* __restrict__ bq, const float* __restrict__ bk,
    const float* __restrict__ bv,
    _Float16* __restrict__ qo, _Float16* __restrict__ ko,
    _Float16* __restrict__ vo) {
  constexpr int LD = 40;  // 32 + 8 pad (halves)
  __shared__ _Float16 As[128 * LD];
  __shared__ _Float16 Bs[128 * LD];

  const int z = blockIdx.z;
  const float* W    = (z == 0) ? Wq : (z == 1) ? Wk : Wv;
  const float* bias = (z == 0) ? bq : (z == 1) ? bk : bv;
  _Float16* out     = (z == 0) ? qo : (z == 1) ? ko : vo;
  const bool phi = (z < 2);

  const int tid = threadIdx.x, lane = tid & 31, w = tid >> 5;
  const int waveM = w & 3, waveN = w >> 2;  // 4 waves M x 2 waves N
  const int m_blk = blockIdx.y * 128, n_blk = blockIdx.x * 128;
  const int srow = tid >> 1, skof = (tid & 1) * 16;

  v8f acc[2][4];
#pragma unroll
  for (int i = 0; i < 2; ++i)
#pragma unroll
    for (int j = 0; j < 4; ++j) acc[i][j] = vzero8();

  for (int k0 = 0; k0 < 1024; k0 += 32) {
    __syncthreads();
    {  // stage A (x) and B (W) as f16
      const float* pa = x + (size_t)(m_blk + srow) * 1024 + k0 + skof;
      const float* pb = W + (size_t)(n_blk + srow) * 1024 + k0 + skof;
      alignas(16) float ta[16], tb[16];
#pragma unroll
      for (int i = 0; i < 4; ++i) {
        *(f4v*)(ta + 4 * i) = *(const f4v*)(pa + 4 * i);
        *(f4v*)(tb + 4 * i) = *(const f4v*)(pb + 4 * i);
      }
      v8h a0, a1, b0, b1;
#pragma unroll
      for (int j = 0; j < 8; ++j) {
        a0[j] = (_Float16)ta[j];     a1[j] = (_Float16)ta[8 + j];
        b0[j] = (_Float16)tb[j];     b1[j] = (_Float16)tb[8 + j];
      }
      *(v8h*)&As[srow * LD + skof]     = a0;
      *(v8h*)&As[srow * LD + skof + 8] = a1;
      *(v8h*)&Bs[srow * LD + skof]     = b0;
      *(v8h*)&Bs[srow * LD + skof + 8] = b1;
    }
    __syncthreads();
    const _Float16* Ab = &As[(waveM * 32) * LD];
    const _Float16* Bb = &Bs[(waveN * 64) * LD];
    v16h a0 = frag_a(Ab, LD, lane);
    v16h a1 = frag_a(Ab + 16 * LD, LD, lane);
#pragma unroll
    for (int nt = 0; nt < 4; ++nt) {
      v16h b = frag_b(Bb + nt * 16 * LD, LD, lane);
      acc[0][nt] = wmma_f16(a0, b, acc[0][nt]);
      acc[1][nt] = wmma_f16(a1, b, acc[1][nt]);
    }
  }

  const int mb = m_blk + waveM * 32, nb = n_blk + waveN * 64;
  const int mro = ((lane >> 4) << 3);
#pragma unroll
  for (int mt = 0; mt < 2; ++mt)
#pragma unroll
    for (int nt = 0; nt < 4; ++nt) {
      const int gn = nb + nt * 16 + (lane & 15);
      const float bb = bias[gn];
      const int head = gn >> 6, dd = gn & 63;
#pragma unroll
      for (int j = 0; j < 8; ++j) {
        const int gm = mb + mt * 16 + j + mro;
        float val = acc[mt][nt][j] + bb;
        if (phi) val = (val > 0.0f) ? (val + 1.0f) : __expf(val);  // elu+1
        const int batch = gm >> 12, sp = gm & 4095;
        const size_t oi =
            (((size_t)(batch * 16 + head) * 4096) + sp) * 64 + dd;
        out[oi] = (_Float16)val;
      }
    }
}

// ---- Kernel 2: chunked causal linear attention, one block per bh ------------
__global__ __launch_bounds__(256) void linattn(
    const _Float16* __restrict__ q, const _Float16* __restrict__ k,
    const _Float16* __restrict__ v, _Float16* __restrict__ o) {
  constexpr int LQ = 72, LT = 136, LS = 136, LST = 72;
  extern __shared__ _Float16 smem[];
  _Float16* qs  = smem;                  // [128][72]  q natural (TDM)
  _Float16* ks  = qs + 128 * LQ;         // [128][72]  k natural (TDM; B of scores)
  _Float16* kts = ks + 128 * LQ;         // [64][136]  k^T        (B of S-upd)
  _Float16* vts = kts + 64 * LT;         // [64][136]  v^T  (B of num1, A of S-upd)
  _Float16* sc  = vts + 64 * LT;         // [128][136] masked scores f16 (A of num1)
  _Float16* st  = sc + 128 * LS;         // [64][72]   S^T f16 (B of num2)
  float* zf  = (float*)(st + 64 * LST);  // [64]  running sum of k
  float* den = zf + 64;                  // [128]

  const int tid = threadIdx.x, lane = tid & 31, w = tid >> 5;
  const int mro = ((lane >> 4) << 3), ncol = lane & 15;
  const size_t base = (size_t)blockIdx.x * 4096 * 64;
  const unsigned qs_off = (unsigned)(size_t)(void*)qs;
  const unsigned ks_off = (unsigned)(size_t)(void*)ks;

  v8f stacc[2];  // persistent S^T accumulators: wave w owns tiles 2w, 2w+1
  stacc[0] = vzero8();
  stacc[1] = vzero8();
  for (int i = tid; i < 64 * LST; i += 256) st[i] = (_Float16)0.0f;
  if (tid < 64) zf[tid] = 0.0f;
  __syncthreads();

  for (int c = 0; c < 32; ++c) {
    // ---- stage: q,k natural tiles via Tensor Data Mover; kT,vT by scatter ----
    {
      if (tid < 32) {  // wave 0 issues both TDM descriptors (EXEC-independent)
        tdm_load_tile_128x64(q + base + (size_t)c * 128 * 64, qs_off);
        tdm_load_tile_128x64(k + base + (size_t)c * 128 * 64, ks_off);
      }
      const int srow = tid >> 1, sdo = (tid & 1) * 32;
      const size_t go = base + (size_t)(c * 128 + srow) * 64 + sdo;
      v8h kv[4], vv[4];
#pragma unroll
      for (int i = 0; i < 4; ++i) {
        kv[i] = *(const v8h*)(k + go + 8 * i);
        vv[i] = *(const v8h*)(v + go + 8 * i);
      }
#pragma unroll
      for (int i = 0; i < 4; ++i)
#pragma unroll
        for (int j = 0; j < 8; ++j) {
          kts[(sdo + 8 * i + j) * LT + srow] = kv[i][j];
          vts[(sdo + 8 * i + j) * LT + srow] = vv[i][j];
        }
      if (c + 1 < 32) {  // warm GL2 for next chunk (global_prefetch_b8)
        __builtin_prefetch(q + go + 128 * 64, 0, 1);
        __builtin_prefetch(k + go + 128 * 64, 0, 1);
        __builtin_prefetch(v + go + 128 * 64, 0, 1);
      }
      if (tid < 32) __builtin_amdgcn_s_wait_tensorcnt(0);
    }
    __syncthreads();

    // ---- phase 1: scores = q k^T (K=64), mask, rowsum -> den, f16 -> sc ----
    const int m0 = w * 16;
    v16h aq0 = frag_a(qs + m0 * LQ, LQ, lane);
    v16h aq1 = frag_a(qs + m0 * LQ + 32, LQ, lane);
    float rs[8];
#pragma unroll
    for (int j = 0; j < 8; ++j) rs[j] = 0.0f;
#pragma unroll
    for (int nt = 0; nt < 8; ++nt) {
      v16h b0 = frag_b(ks + nt * 16 * LQ, LQ, lane);
      v16h b1 = frag_b(ks + nt * 16 * LQ + 32, LQ, lane);
      v8f s = vzero8();
      s = wmma_f16(aq0, b0, s);
      s = wmma_f16(aq1, b1, s);
#pragma unroll
      for (int j = 0; j < 8; ++j) {
        const int m = m0 + j + mro, n = nt * 16 + ncol;
        const float val = (m >= n) ? s[j] : 0.0f;  // causal mask
        rs[j] += val;
        sc[m * LS + n] = (_Float16)val;
      }
    }
#pragma unroll
    for (int j = 0; j < 8; ++j) {  // reduce over 16-lane group (same row)
      float r = rs[j];
      r += __shfl_xor(r, 1);
      r += __shfl_xor(r, 2);
      r += __shfl_xor(r, 4);
      r += __shfl_xor(r, 8);
      if ((lane & 15) == 0) den[m0 + j + mro] = r;
    }
    __syncthreads();
    if (tid < 128) {  // den += q . z  (z from previous chunks)
      float a = 0.0f;
#pragma unroll
      for (int d = 0; d < 64; ++d) a += (float)qs[tid * LQ + d] * zf[d];
      den[tid] += a;
    }
    __syncthreads();

    // ---- phase 2: num = sc @ v + q @ S_old ; out = num / den ----
    v16h as0 = frag_a(sc + m0 * LS + 0, LS, lane);
    v16h as1 = frag_a(sc + m0 * LS + 32, LS, lane);
    v16h as2 = frag_a(sc + m0 * LS + 64, LS, lane);
    v16h as3 = frag_a(sc + m0 * LS + 96, LS, lane);
#pragma unroll
    for (int nt = 0; nt < 4; ++nt) {
      v8f acc = vzero8();
      v16h bs0 = frag_b(st + nt * 16 * LST, LST, lane);
      v16h bs1 = frag_b(st + nt * 16 * LST + 32, LST, lane);
      acc = wmma_f16(aq0, bs0, acc);  // q @ S^T rows
      acc = wmma_f16(aq1, bs1, acc);
      v16h bv0 = frag_b(vts + nt * 16 * LT + 0, LT, lane);
      v16h bv1 = frag_b(vts + nt * 16 * LT + 32, LT, lane);
      v16h bv2 = frag_b(vts + nt * 16 * LT + 64, LT, lane);
      v16h bv3 = frag_b(vts + nt * 16 * LT + 96, LT, lane);
      acc = wmma_f16(as0, bv0, acc);
      acc = wmma_f16(as1, bv1, acc);
      acc = wmma_f16(as2, bv2, acc);
      acc = wmma_f16(as3, bv3, acc);
#pragma unroll
      for (int j = 0; j < 8; ++j) {
        const int m = m0 + j + mro, e = nt * 16 + ncol;
        const float val = acc[j] / den[m];
        o[base + (size_t)(c * 128 + m) * 64 + e] = (_Float16)val;
      }
    }
    __syncthreads();  // st is read above, rewritten below

    // ---- phase 3: S^T += v^T k (K=128), z += sum_r k, refresh st f16 ----
#pragma unroll
    for (int ti = 0; ti < 2; ++ti) {
      const int tt = w * 2 + ti, mt = tt >> 2, nt = tt & 3;
      v16h av0 = frag_a(vts + mt * 16 * LT + 0, LT, lane);
      v16h av1 = frag_a(vts + mt * 16 * LT + 32, LT, lane);
      v16h av2 = frag_a(vts + mt * 16 * LT + 64, LT, lane);
      v16h av3 = frag_a(vts + mt * 16 * LT + 96, LT, lane);
      v16h bk0 = frag_b(kts + nt * 16 * LT + 0, LT, lane);
      v16h bk1 = frag_b(kts + nt * 16 * LT + 32, LT, lane);
      v16h bk2 = frag_b(kts + nt * 16 * LT + 64, LT, lane);
      v16h bk3 = frag_b(kts + nt * 16 * LT + 96, LT, lane);
      stacc[ti] = wmma_f16(av0, bk0, stacc[ti]);
      stacc[ti] = wmma_f16(av1, bk1, stacc[ti]);
      stacc[ti] = wmma_f16(av2, bk2, stacc[ti]);
      stacc[ti] = wmma_f16(av3, bk3, stacc[ti]);
    }
    if (tid < 64) {
      float a = 0.0f;
      for (int r = 0; r < 128; ++r) a += (float)kts[tid * LT + r];
      zf[tid] += a;
    }
#pragma unroll
    for (int ti = 0; ti < 2; ++ti) {
      const int tt = w * 2 + ti, mt = tt >> 2, nt = tt & 3;
#pragma unroll
      for (int j = 0; j < 8; ++j) {
        const int e = mt * 16 + j + mro, d = nt * 16 + ncol;
        st[e * LST + d] = (_Float16)stacc[ti][j];
      }
    }
    __syncthreads();
  }
}

// ---- Kernel 3: output projection, f16 head-major attn * Wo^T + bo -> f32 ----
__global__ __launch_bounds__(256) void out_gemm(
    const _Float16* __restrict__ a, const float* __restrict__ Wo,
    const float* __restrict__ bo, float* __restrict__ y) {
  constexpr int LD = 40;
  __shared__ _Float16 As[128 * LD];
  __shared__ _Float16 Bs[128 * LD];

  const int tid = threadIdx.x, lane = tid & 31, w = tid >> 5;
  const int waveM = w & 3, waveN = w >> 2;
  const int m_blk = blockIdx.y * 128, n_blk = blockIdx.x * 128;
  const int srow = tid >> 1, skof = (tid & 1) * 16;

  v8f acc[2][4];
#pragma unroll
  for (int i = 0; i < 2; ++i)
#pragma unroll
    for (int j = 0; j < 4; ++j) acc[i][j] = vzero8();

  for (int k0 = 0; k0 < 1024; k0 += 32) {
    __syncthreads();
    {  // A: gather head-major f16; B: Wo f32 -> f16
      const int gm = m_blk + srow, kk = k0 + skof;
      const int head = kk >> 6, batch = gm >> 12, sp = gm & 4095;
      const _Float16* pa =
          a + (((size_t)(batch * 16 + head) * 4096) + sp) * 64 + (kk & 63);
      *(v8h*)&As[srow * LD + skof]     = *(const v8h*)pa;
      *(v8h*)&As[srow * LD + skof + 8] = *(const v8h*)(pa + 8);

      const float* pb = Wo + (size_t)(n_blk + srow) * 1024 + k0 + skof;
      alignas(16) float tb[16];
#pragma unroll
      for (int i = 0; i < 4; ++i) *(f4v*)(tb + 4 * i) = *(const f4v*)(pb + 4 * i);
      v8h b0, b1;
#pragma unroll
      for (int j = 0; j < 8; ++j) {
        b0[j] = (_Float16)tb[j];
        b1[j] = (_Float16)tb[8 + j];
      }
      *(v8h*)&Bs[srow * LD + skof]     = b0;
      *(v8h*)&Bs[srow * LD + skof + 8] = b1;
    }
    __syncthreads();
    const _Float16* Ab = &As[(waveM * 32) * LD];
    const _Float16* Bb = &Bs[(waveN * 64) * LD];
    v16h a0 = frag_a(Ab, LD, lane);
    v16h a1 = frag_a(Ab + 16 * LD, LD, lane);
#pragma unroll
    for (int nt = 0; nt < 4; ++nt) {
      v16h b = frag_b(Bb + nt * 16 * LD, LD, lane);
      acc[0][nt] = wmma_f16(a0, b, acc[0][nt]);
      acc[1][nt] = wmma_f16(a1, b, acc[1][nt]);
    }
  }

  const int mb = m_blk + waveM * 32, nb = n_blk + waveN * 64;
  const int mro = ((lane >> 4) << 3);
#pragma unroll
  for (int mt = 0; mt < 2; ++mt)
#pragma unroll
    for (int nt = 0; nt < 4; ++nt) {
      const int gn = nb + nt * 16 + (lane & 15);
      const float bb = bo[gn];
#pragma unroll
      for (int j = 0; j < 8; ++j) {
        const int gm = mb + mt * 16 + j + mro;
        y[(size_t)gm * 1024 + gn] = acc[mt][nt][j] + bb;
      }
    }
}

extern "C" void kernel_launch(void* const* d_in, const int* in_sizes, int n_in,
                              void* d_out, int out_size, void* d_ws,
                              size_t ws_size, hipStream_t stream) {
  (void)in_sizes; (void)n_in; (void)out_size; (void)ws_size;
  const float* x  = (const float*)d_in[0];
  const float* Wq = (const float*)d_in[1];
  const float* bq = (const float*)d_in[2];
  const float* Wk = (const float*)d_in[3];
  const float* bk = (const float*)d_in[4];
  const float* Wv = (const float*)d_in[5];
  const float* bv = (const float*)d_in[6];
  const float* Wo = (const float*)d_in[7];
  const float* bo = (const float*)d_in[8];
  float* out = (float*)d_out;

  const size_t E = (size_t)16384 * 1024;  // per-tensor elements
  _Float16* qws = (_Float16*)d_ws;
  _Float16* kws = qws + E;
  _Float16* vws = kws + E;
  _Float16* aws = vws + E;

  // 1) fused QKV projections with phi epilogue on q,k
  qkv_gemm<<<dim3(8, 128, 3), 256, 0, stream>>>(x, Wq, Wk, Wv, bq, bk, bv,
                                                qws, kws, vws);

  // 2) chunked causal linear attention (one WG per batch*head)
  constexpr size_t SMEM =
      (size_t)(128 * 72 + 128 * 72 + 64 * 136 + 64 * 136 + 128 * 136 +
               64 * 72) * sizeof(_Float16) + (64 + 128) * sizeof(float);
  (void)hipFuncSetAttribute((const void*)linattn,
                            hipFuncAttributeMaxDynamicSharedMemorySize,
                            (int)SMEM);
  linattn<<<dim3(64), 256, SMEM, stream>>>(qws, kws, vws, aws);

  // 3) output projection
  out_gemm<<<dim3(8, 128), 256, 0, stream>>>(aws, Wo, bo, out);
}
